// MemoryBank_17179869184010
// MI455X (gfx1250) — matrix operand order repaired and checked by reading, not compiled, not run on hardware
//
#include <hip/hip_runtime.h>
#include <math.h>

#define D      1024
#define TILE   16          // WMMA tile edge
#define MSUB   2           // M sub-tiles per block
#define MROWS  (TILE*MSUB) // 32 query rows per block
#define WAVES  8
#define NEG_INF (-__builtin_inff())

typedef __attribute__((ext_vector_type(16))) __bf16         v16bf;
typedef __attribute__((ext_vector_type(8)))  float          v8f;
typedef __attribute__((ext_vector_type(8)))  unsigned short v8us;

union Frag { v8us h[2]; v16bf b; };

// float -> bf16 round-to-nearest-even, as raw ushort
__device__ __forceinline__ unsigned short f2bf(float f) {
    unsigned u = __float_as_uint(f);
    u += 0x7FFFu + ((u >> 16) & 1u);
    return (unsigned short)(u >> 16);
}

// One wave32 per row: L2-normalize a 1024-wide f32 row, emit bf16 (ushort) row.
__global__ void l2norm_bf16_kernel(const float* __restrict__ src,
                                   unsigned short* __restrict__ dst,
                                   int rows) {
    const int gw   = (blockIdx.x * blockDim.x + threadIdx.x) >> 5;
    const int lane = threadIdx.x & 31;
    if (gw >= rows) return;

    const float4* r = (const float4*)(src + (size_t)gw * D);
    float4 v[8];
    float s = 0.f;
#pragma unroll
    for (int c = 0; c < 8; ++c) {
        v[c] = r[c * 32 + lane];
        s += v[c].x * v[c].x + v[c].y * v[c].y + v[c].z * v[c].z + v[c].w * v[c].w;
    }
#pragma unroll
    for (int off = 16; off; off >>= 1) s += __shfl_xor(s, off);
    const float scale = 1.f / fmaxf(sqrtf(s), 1e-12f);

    unsigned short* o = dst + (size_t)gw * D;
#pragma unroll
    for (int c = 0; c < 8; ++c) {
        ushort4 p;
        p.x = f2bf(v[c].x * scale);
        p.y = f2bf(v[c].y * scale);
        p.z = f2bf(v[c].z * scale);
        p.w = f2bf(v[c].w * scale);
        *(ushort4*)(o + c * 128 + lane * 4) = p;
    }
}

// Fused: sim GEMM (bf16 WMMA, f32 acc) -> running top-16 -> merge -> weighted gather.
// One block = 32 query rows (2 sub-tiles); each B fragment feeds 2 WMMAs (L2 reuse x2).
__global__ void __launch_bounds__(256)
sim_topk_gather_kernel(const unsigned short* __restrict__ qn,
                       const unsigned short* __restrict__ mn,
                       const float* __restrict__ memory,
                       float* __restrict__ out,
                       int ntiles) {
    __shared__ unsigned short qtile[MROWS][D + 8];     // padded stride vs bank conflicts
    __shared__ float          tv [WAVES][MROWS][16];   // per-wave top-16 values
    __shared__ unsigned short ti [WAVES][MROWS][16];   // per-wave top-16 memory indices
    __shared__ float          thr[WAVES][MROWS];       // running 16th-best (threshold)
    __shared__ float          fw [MROWS][16];          // merged weights
    __shared__ int            fi [MROWS][16];          // merged indices

    const int tid   = threadIdx.x;
    const int w     = tid >> 5;
    const int lane  = tid & 31;
    const int half  = lane >> 4;
    const int l15   = lane & 15;
    const int qbase = blockIdx.x * MROWS;

    // Stage normalized bf16 query rows into LDS (16B vector copies).
    for (int i = tid; i < MROWS * (D / 8); i += 256) {
        const int row = i >> 7;
        const int c8  = (i & 127) * 8;
        *(v8us*)&qtile[row][c8] = *(const v8us*)(qn + (size_t)(qbase + row) * D + c8);
    }
    // Init per-wave top-k state.
    for (int i = lane; i < MROWS * 16; i += 32) { tv[w][i >> 4][i & 15] = NEG_INF; ti[w][i >> 4][i & 15] = 0; }
    for (int i = lane; i < MROWS; i += 32) thr[w][i] = NEG_INF;
    __syncthreads();

    // ---- main streaming GEMM + top-k ----
    for (int nb = w; nb < ntiles; nb += WAVES) {
        const int nbase = nb * TILE;
        v8f acc[MSUB];
        acc[0] = (v8f){0.f,0.f,0.f,0.f,0.f,0.f,0.f,0.f};
        acc[1] = (v8f){0.f,0.f,0.f,0.f,0.f,0.f,0.f,0.f};

        // B fragment source: lane's memory row, same K interleave as A.
        const unsigned short* brow = mn + (size_t)(nbase + l15) * D;
        // Prefetch next n-tile's B row (lowers to global_prefetch_b8).
        if (nb + WAVES < ntiles)
            __builtin_prefetch(mn + (size_t)(nbase + WAVES * TILE + l15) * D, 0, 0);

#pragma unroll 2
        for (int kb = 0; kb < D; kb += 32) {
            Frag A0, A1, Bv;
            Bv.h[0] = *(const v8us*)(brow + kb + half * 8);
            Bv.h[1] = *(const v8us*)(brow + kb + half * 8 + 16);
            // A layout: lanes 0-15 hold K {kb..kb+7, kb+16..kb+23}; lanes 16-31 +8.
            A0.h[0] = *(const v8us*)&qtile[l15][kb + half * 8];
            A0.h[1] = *(const v8us*)&qtile[l15][kb + half * 8 + 16];
            A1.h[0] = *(const v8us*)&qtile[l15 + TILE][kb + half * 8];
            A1.h[1] = *(const v8us*)&qtile[l15 + TILE][kb + half * 8 + 16];
            acc[0] = __builtin_amdgcn_wmma_f32_16x16x32_bf16(false, A0.b, false, Bv.b,
                                                             (short)0, acc[0], false, false);
            acc[1] = __builtin_amdgcn_wmma_f32_16x16x32_bf16(false, A1.b, false, Bv.b,
                                                             (short)0, acc[1], false, false);
        }

        // Candidate filtering: only values beating the running 16th get inserted.
        const int midx = nbase + l15;
#pragma unroll
        for (int s = 0; s < MSUB; ++s) {
#pragma unroll
            for (int r = 0; r < 8; ++r) {
                const float v   = acc[s][r];
                const int   row = s * TILE + r + half * 8;   // C layout: VGPR r -> M=r / M=r+8
                const bool cand = (v > thr[w][row]) && (v != 1.0f);  // mask exact matches
                unsigned mask = (unsigned)__ballot(cand);
                while (mask) {
                    const int src = __ffs(mask) - 1;
                    mask &= mask - 1;
                    const float cv   = __shfl(v, src);
                    const int   crow = __shfl(row, src);
                    const int   cidx = __shfl(midx, src);
                    if (lane == 0) {
                        float mnv = tv[w][crow][0]; int mp = 0;
                        for (int s2 = 1; s2 < 16; ++s2) {
                            const float t = tv[w][crow][s2];
                            if (t < mnv) { mnv = t; mp = s2; }
                        }
                        if (cv > mnv) {
                            tv[w][crow][mp] = cv;
                            ti[w][crow][mp] = (unsigned short)cidx;
                            float nt = tv[w][crow][0];
                            for (int s2 = 1; s2 < 16; ++s2) nt = fminf(nt, tv[w][crow][s2]);
                            thr[w][crow] = nt;
                        }
                    }
                }
            }
        }
    }
    __syncthreads();

    // ---- merge 8 per-wave lists -> global top-16 per row (waves 0..1, 16 rows each) ----
    if (w < MSUB) {
        for (int rr = 0; rr < TILE; ++rr) {
            const int row = w * TILE + rr;
            float cv[4]; int ci[4];
#pragma unroll
            for (int j = 0; j < 4; ++j) {
                const int c = lane * 4 + j;           // 128 candidates, 4 per lane
                cv[j] = tv[c >> 4][row][c & 15];
                ci[j] = ti[c >> 4][row][c & 15];
            }
            for (int sel = 0; sel < 16; ++sel) {
                float bv = cv[0]; int bj = 0;
#pragma unroll
                for (int j = 1; j < 4; ++j) if (cv[j] > bv) { bv = cv[j]; bj = j; }
                int bc = lane * 4 + bj;
#pragma unroll
                for (int off = 16; off; off >>= 1) {
                    const float ov = __shfl_xor(bv, off);
                    const int   oc = __shfl_xor(bc, off);
                    if (ov > bv || (ov == bv && oc < bc)) { bv = ov; bc = oc; }
                }
                const int wi = __shfl(ci[bc & 3], bc >> 2);  // winner's memory index
                if ((bc >> 2) == lane) cv[bc & 3] = NEG_INF; // consume winner
                if (lane == 0) {
                    fw[row][sel] = __builtin_isfinite(bv) ? bv : 0.f;  // zero masked slots
                    fi[row][sel] = wi;
                }
            }
        }
    }
    __syncthreads();

    // ---- weighted gather from ORIGINAL f32 memory rows ----
    const int d0 = tid * 4;                           // 256 threads x float4 = 1024 dims
    for (int row = 0; row < MROWS; ++row) {
        float4 accv = make_float4(0.f, 0.f, 0.f, 0.f);
#pragma unroll
        for (int k = 0; k < 16; ++k) {
            const float wk = fw[row][k];
            if (wk != 0.f) {
                const float4 mv = *(const float4*)(memory + (size_t)fi[row][k] * D + d0);
                accv.x += wk * mv.x; accv.y += wk * mv.y;
                accv.z += wk * mv.z; accv.w += wk * mv.w;
            }
        }
        *(float4*)(out + (size_t)(qbase + row) * D + d0) = accv;
    }
}

extern "C" void kernel_launch(void* const* d_in, const int* in_sizes, int n_in,
                              void* d_out, int out_size, void* d_ws, size_t ws_size,
                              hipStream_t stream) {
    const float* q   = (const float*)d_in[0];
    const float* m   = (const float*)d_in[1];
    float*       out = (float*)d_out;

    const int B = in_sizes[0] / D;   // 4096 query rows
    const int M = in_sizes[1] / D;   // 8192 memory rows

    // workspace: normalized bf16 query (B*D) then memory (M*D)
    unsigned short* qn  = (unsigned short*)d_ws;
    unsigned short* mnp = qn + (size_t)B * D;

    l2norm_bf16_kernel<<<(B * 32 + 255) / 256, 256, 0, stream>>>(q, qn, B);
    l2norm_bf16_kernel<<<(M * 32 + 255) / 256, 256, 0, stream>>>(m, mnp, M);

    sim_topk_gather_kernel<<<B / MROWS, 256, 0, stream>>>(qn, mnp, m, out, M / TILE);
}
